// VectorQuantizer_reset_38242388804088
// MI455X (gfx1250) — matrix operand order repaired and checked by reading, not compiled, not run on hardware
//
#include <hip/hip_runtime.h>
#include <hip/hip_bf16.h>
#include <stdint.h>

// ---------------------------------------------------------------------------
// VQ-VAE vector quantizer for MI455X (gfx1250).
//   d = ||z||^2 + ||e||^2 - 2 z.e   -> argmin over 1024 codes (||z||^2 dropped)
//   Distance ranking via v_wmma_f32_16x16x32_bf16 (f32 accumulate).
//   Codebook chunks double-buffered into LDS with global_load_async_to_lds.
//   z_q gather / loss computed from the original f32 tables.
// ---------------------------------------------------------------------------

typedef __attribute__((ext_vector_type(16))) __bf16          v16bf;
typedef __attribute__((ext_vector_type(16))) unsigned short  v16u;
typedef __attribute__((ext_vector_type(8)))  float           v8f;
typedef __attribute__((ext_vector_type(4)))  int             v4i;

union BfPack { v16u us; v16bf bf; };

#define VQ_EDIM 256
#define VQ_ROWS_PER_BLOCK 128   // 8 waves * 16 rows
#define VQ_CHUNK 16             // codes per WMMA N tile

static __device__ __host__ inline unsigned short f2bfbits(float f) {
  union { float f; unsigned u; } v; v.f = f;
  unsigned u = v.u;
  unsigned r = u + 0x7FFFu + ((u >> 16) & 1u);   // round-to-nearest-even
  return (unsigned short)(r >> 16);
}

#if defined(__AMDGCN__) && __has_builtin(__builtin_amdgcn_global_load_async_to_lds_b128)
#define VQ_HAVE_ASYNC 1
typedef __attribute__((address_space(1))) v4i* gptr_v4i;
typedef __attribute__((address_space(3))) v4i* lptr_v4i;
#endif

// Stage one chunk (16 code rows of bf16 = 8192 B) into LDS; 256 threads x 32 B.
static __device__ inline void vq_stage(const __bf16* __restrict__ ebf,
                                       __bf16* lb, int j0, int t) {
  __bf16* g = const_cast<__bf16*>(ebf) + (size_t)j0 * VQ_EDIM + t * 16;
  __bf16* l = lb + t * 16;
#if defined(VQ_HAVE_ASYNC)
  __builtin_amdgcn_global_load_async_to_lds_b128(
      (gptr_v4i)(void*)g, (lptr_v4i)(void*)l, 0, 0);
  __builtin_amdgcn_global_load_async_to_lds_b128(
      (gptr_v4i)(void*)(g + 8), (lptr_v4i)(void*)(l + 8), 0, 0);
#else
  const uint4* gv = (const uint4*)g;
  uint4 a = gv[0], b = gv[1];
  ((uint4*)l)[0] = a;
  ((uint4*)l)[1] = b;
#endif
}

static __device__ inline void vq_wait_async() {
#if defined(VQ_HAVE_ASYNC)
#if __has_builtin(__builtin_amdgcn_s_wait_asynccnt)
  __builtin_amdgcn_s_wait_asynccnt(0);
#else
  asm volatile("s_wait_asynccnt 0" ::: "memory");
#endif
#endif
}

// --------------------------------------------------------------------------
// Kernel 1: emb(f32) -> bf16 copy + ||e_j||^2, zero loss accumulator.
// grid = n_e blocks, block = 32 (one wave per code row).
// --------------------------------------------------------------------------
__global__ void vq_prep(const float* __restrict__ emb, __bf16* __restrict__ ebf,
                        float* __restrict__ norms, float* __restrict__ lacc) {
  const int j = blockIdx.x;
  const int lane = threadIdx.x;
  const float* er = emb + (size_t)j * VQ_EDIM;
  float s = 0.f;
#pragma unroll
  for (int part = 0; part < 2; ++part) {
    int off = part * 128 + lane * 4;
    float4 v = *(const float4*)(er + off);
    uint2 o;
    o.x = (unsigned)f2bfbits(v.x) | ((unsigned)f2bfbits(v.y) << 16);
    o.y = (unsigned)f2bfbits(v.z) | ((unsigned)f2bfbits(v.w) << 16);
    *(uint2*)(ebf + (size_t)j * VQ_EDIM + off) = o;
    s += v.x * v.x + v.y * v.y + v.z * v.z + v.w * v.w;
  }
#pragma unroll
  for (int m = 16; m >= 1; m >>= 1) s += __shfl_xor(s, m, 32);
  if (lane == 0) {
    norms[j] = s;
    if (j == 0) lacc[0] = 0.f;
  }
}

// --------------------------------------------------------------------------
// Kernel 2: distances via WMMA, argmin, gather z_q, partial loss.
// grid = N/128 blocks, block = 256 (8 waves; wave w owns rows w*16..w*16+15).
// --------------------------------------------------------------------------
__global__ __launch_bounds__(256) void vq_main(
    const float* __restrict__ z, const float* __restrict__ emb,
    const __bf16* __restrict__ ebf, const float* __restrict__ norms,
    float* __restrict__ zq, float* __restrict__ idxf,
    float* __restrict__ lacc, int ne) {
  __shared__ __attribute__((aligned(32))) __bf16 lbuf[2][VQ_CHUNK * VQ_EDIM];
  __shared__ int s_idx[VQ_ROWS_PER_BLOCK];

  const int t = threadIdx.x;
  const int w = t >> 5;
  const int lane = t & 31;
  const int hi = lane >> 4;     // lane half: selects K sub-chunks / M half
  const int nlo = lane & 15;    // N index within tile / M row for A frags

  const long long row0 = (long long)blockIdx.x * VQ_ROWS_PER_BLOCK + w * 16;
  const long long arow = row0 + nlo;   // this lane's A-matrix row (M = lane%16)

  // ---- Build all 8 A fragments (K = 0..255) in registers, f32 -> bf16. ----
  // A k order used here (contiguous, same as B): lane<16 holds K {kk*32+0..15},
  //                        lane>=16 holds K {kk*32+16..31}.
  v16bf afr[8];
  {
    const float* rp = z + arow * VQ_EDIM;
#pragma unroll
    for (int kk = 0; kk < 8; ++kk) {
      const float4* q0 = (const float4*)(rp + kk * 32 + hi * 16);
      const float4* q1 = (const float4*)(rp + kk * 32 + hi * 16 + 8);
      float4 x0 = q0[0], x1 = q0[1], x2 = q1[0], x3 = q1[1];
      BfPack u;
      u.us[0]  = f2bfbits(x0.x); u.us[1]  = f2bfbits(x0.y);
      u.us[2]  = f2bfbits(x0.z); u.us[3]  = f2bfbits(x0.w);
      u.us[4]  = f2bfbits(x1.x); u.us[5]  = f2bfbits(x1.y);
      u.us[6]  = f2bfbits(x1.z); u.us[7]  = f2bfbits(x1.w);
      u.us[8]  = f2bfbits(x2.x); u.us[9]  = f2bfbits(x2.y);
      u.us[10] = f2bfbits(x2.z); u.us[11] = f2bfbits(x2.w);
      u.us[12] = f2bfbits(x3.x); u.us[13] = f2bfbits(x3.y);
      u.us[14] = f2bfbits(x3.z); u.us[15] = f2bfbits(x3.w);
      afr[kk] = u.bf;
    }
  }

  float minv[8];
  int   mini[8];
#pragma unroll
  for (int r = 0; r < 8; ++r) { minv[r] = 3.4e38f; mini[r] = 0; }

  const int nch = ne >> 4;  // chunks of 16 codes

  // Prologue: stage chunk 0, then double-buffer through all chunks.
  vq_stage(ebf, &lbuf[0][0], 0, t);
  vq_wait_async();
  __syncthreads();

  for (int jc = 0; jc < nch; ++jc) {
    if (jc + 1 < nch) vq_stage(ebf, &lbuf[(jc + 1) & 1][0], (jc + 1) * VQ_CHUNK, t);

    const __bf16* lb = &lbuf[jc & 1][0];
    v8f c = {0.f, 0.f, 0.f, 0.f, 0.f, 0.f, 0.f, 0.f};
#pragma unroll
    for (int kk = 0; kk < 8; ++kk) {
      // B 32x16 layout: lane n holds E[code j0+n][kbase .. kbase+15],
      // kbase = kk*32 + hi*16  -> contiguous 32-byte LDS read.
      v16bf b = *(const v16bf*)(lb + nlo * VQ_EDIM + kk * 32 + hi * 16);
      c = __builtin_amdgcn_wmma_f32_16x16x32_bf16(
          false, afr[kk], false, b, (short)0, c, false, false);
    }

    const int   code = jc * VQ_CHUNK + nlo;
    const float en   = norms[code];
#pragma unroll
    for (int r = 0; r < 8; ++r) {
      float sd = en - 2.0f * c[r];       // ||e||^2 - 2 z.e  (||z||^2 dropped)
      if (sd < minv[r]) { minv[r] = sd; mini[r] = code; }
    }

    vq_wait_async();   // next buffer resident before anyone reads it
    __syncthreads();   // everyone done with current buffer before overwrite
  }

  // ---- Reduce argmin across the 16-lane N groups (C/D: lanes 0-15 -> M=r,
  //      lanes 16-31 -> M=8+r). Tie-break to lowest index = argmin semantics.
#pragma unroll
  for (int r = 0; r < 8; ++r) {
    float v = minv[r];
    int   i = mini[r];
#pragma unroll
    for (int m = 1; m < 16; m <<= 1) {
      float ov = __shfl_xor(v, m, 32);
      int   oi = __shfl_xor(i, m, 32);
      if (ov < v || (ov == v && oi < i)) { v = ov; i = oi; }
    }
    minv[r] = v; mini[r] = i;
  }
  if (nlo == 0) {
#pragma unroll
    for (int r = 0; r < 8; ++r) s_idx[w * 16 + hi * 8 + r] = mini[r];
  }
  __syncthreads();

  // ---- Phase 2: coalesced gather of z_q, index output, partial loss. ----
  float ls = 0.f;
  for (int m = 0; m < 16; ++m) {
    long long row = row0 + m;
    int idx = s_idx[w * 16 + m];
    const float* zr = z   + row * VQ_EDIM;
    const float* er = emb + (long long)idx * VQ_EDIM;
    float*       qo = zq  + row * VQ_EDIM;
#pragma unroll
    for (int part = 0; part < 2; ++part) {
      int off = part * 128 + lane * 4;
      float4 zv = *(const float4*)(zr + off);
      float4 ev = *(const float4*)(er + off);
      *(float4*)(qo + off) = ev;          // z_q_st forward value == z_q
      float dx = ev.x - zv.x, dy = ev.y - zv.y;
      float dz = ev.z - zv.z, dw = ev.w - zv.w;
      ls += dx * dx + dy * dy + dz * dz + dw * dw;
    }
    if (lane == 0) idxf[row] = (float)idx;
  }
#pragma unroll
  for (int m = 16; m >= 1; m >>= 1) ls += __shfl_xor(ls, m, 32);
  if (lane == 0) atomicAdd(lacc, ls);
}

// --------------------------------------------------------------------------
// Kernel 3: loss = (1 + BETA) * sum / (N * 256)
// --------------------------------------------------------------------------
__global__ void vq_fin(const float* __restrict__ lacc, float* __restrict__ loss,
                       long long n) {
  loss[0] = 1.25f * lacc[0] / (float)(n * VQ_EDIM);
}

// --------------------------------------------------------------------------
extern "C" void kernel_launch(void* const* d_in, const int* in_sizes, int n_in,
                              void* d_out, int out_size, void* d_ws, size_t ws_size,
                              hipStream_t stream) {
  const float* z   = (const float*)d_in[0];
  const float* emb = (const float*)d_in[1];
  const long long n  = in_sizes[0] / VQ_EDIM;   // 131072
  const int       ne = in_sizes[1] / VQ_EDIM;   // 1024

  float* zq   = (float*)d_out;                  // [n, 256]
  float* loss = zq + (size_t)n * VQ_EDIM;       // [1]
  float* idxf = loss + 1;                       // [n] (indices as float)

  __bf16* ebf  = (__bf16*)d_ws;                                     // 512 KB
  float* norms = (float*)((char*)d_ws + (size_t)ne * VQ_EDIM * 2);  // 4 KB
  float* lacc  = norms + ne;                                        // 4 B

  vq_prep<<<ne, 32, 0, stream>>>(emb, ebf, norms, lacc);
  vq_main<<<(int)(n / VQ_ROWS_PER_BLOCK), 256, 0, stream>>>(
      z, emb, ebf, norms, zq, idxf, lacc, ne);
  vq_fin<<<1, 1, 0, stream>>>(lacc, loss, n);
}
